// BiGRU_56831007261329
// MI455X (gfx1250) — matrix operand order repaired and checked by reading, not compiled, not run on hardware
//
#include <hip/hip_runtime.h>
#include <hip/hip_bf16.h>

// Problem sizes (fixed by the reference)
#define NB   64            // batch
#define TT   512           // time steps
#define DD   512           // input dim
#define HH   512           // hidden dim
#define H3   1536          // 3*H
#define MM   (NB * TT)     // rows of input-projection GEMM = 32768

typedef __attribute__((ext_vector_type(16))) __bf16 v16bf;
typedef __attribute__((ext_vector_type(8)))  __bf16 v8bf;
typedef __attribute__((ext_vector_type(8)))  float  v8f;

// ---------- bf16 helpers (bit-exact RNE, no reliance on __bf16 arithmetic) ----
__device__ __forceinline__ unsigned short f2bf(float f) {
    union { float f; unsigned u; } v; v.f = f;
    unsigned r = v.u + 0x7FFFu + ((v.u >> 16) & 1u);
    return (unsigned short)(r >> 16);
}
__device__ __forceinline__ float bf2f(unsigned short h) {
    union { unsigned u; float f; } v; v.u = ((unsigned)h) << 16;
    return v.f;
}
__device__ __forceinline__ float sigmoidf(float x) {
    return 1.0f / (1.0f + __expf(-x));
}

// Build a 16-bit A fragment: lane holds row (lane&15), K chunks at
// kbase..kbase+7 and kbase+16..kbase+23 where kbase = (lane>>4)*8 (ISA 7.12.2).
__device__ __forceinline__ v16bf load_a_frag(const unsigned short* p) {
    v8bf lo = *(const v8bf*)(p);
    v8bf hi = *(const v8bf*)(p + 16);
    v16bf a;
#pragma unroll
    for (int i = 0; i < 8; ++i) { a[i] = lo[i]; a[i + 8] = hi[i]; }
    return a;
}

// ------------------------------- init ----------------------------------------
__global__ void init_barriers(unsigned* bars) {
    if (threadIdx.x < 2) bars[threadIdx.x] = 0u;
}

// fp32 x -> bf16 x (same layout), 4 elements/thread
__global__ void convert_x_kernel(const float* __restrict__ x,
                                 unsigned short* __restrict__ xbf) {
    int i = (blockIdx.x * blockDim.x + threadIdx.x) * 4;
#pragma unroll
    for (int k = 0; k < 4; ++k) xbf[i + k] = f2bf(x[i + k]);
}

// Transpose+convert the four weight matrices: in fp32 [D][3H] -> out bf16 [3H][D]
__global__ void transpose_w_kernel(const float* __restrict__ Wxf,
                                   const float* __restrict__ Whf,
                                   const float* __restrict__ Wxb,
                                   const float* __restrict__ Whb,
                                   unsigned short* __restrict__ wxT,
                                   unsigned short* __restrict__ whT) {
    const float* src;
    unsigned short* dst;
    switch (blockIdx.y) {
        case 0: src = Wxf; dst = wxT;                 break;
        case 1: src = Wxb; dst = wxT + H3 * DD;       break;
        case 2: src = Whf; dst = whT;                 break;
        default: src = Whb; dst = whT + H3 * DD;      break;
    }
    int idx = blockIdx.x * blockDim.x + threadIdx.x;  // 0 .. 3H*D-1
    int r = idx / DD;        // output row   (3H index)
    int c = idx - r * DD;    // output col   (D index)
    dst[r * DD + c] = f2bf(src[c * H3 + r]);
}

// ---------------- input projection GEMM: xm = x(_flipped) @ W_x --------------
// grid (MM/64, H3/64, 2), block 128 (4 waves). Wave w owns rows m0+16w,
// 4 C tiles of 16x16 across a 64-col strip. K = 512 in chunks of 32.
__global__ void proj_gemm_kernel(const unsigned short* __restrict__ xbf,
                                 const unsigned short* __restrict__ wxT,
                                 float* __restrict__ xm) {
    const int dir  = blockIdx.z;
    const int lane = threadIdx.x & 31;
    const int wave = threadIdx.x >> 5;
    const int m0   = blockIdx.x * 64 + wave * 16;
    const int n0   = blockIdx.y * 64;
    const int lo   = lane & 15;
    const int hi   = lane >> 4;

    // Per-lane A source row (time-flipped for backward direction)
    int m = m0 + lo;
    int n = m >> 9;            // batch (T = 512)
    int t = m & 511;
    if (dir) t = (TT - 1) - t;
    const unsigned short* arow = xbf + (size_t)((n << 9) | t) * DD + hi * 8;

    const unsigned short* B = wxT + (size_t)dir * H3 * DD;
    float* out = xm + (size_t)dir * MM * H3;

    v8f acc[4];
#pragma unroll
    for (int ct = 0; ct < 4; ++ct) acc[ct] = (v8f)(0.0f);

#pragma unroll 4
    for (int kc = 0; kc < DD / 32; ++kc) {
        v16bf a = load_a_frag(arow + kc * 32);
#pragma unroll
        for (int ct = 0; ct < 4; ++ct) {
            const unsigned short* bp =
                B + (size_t)(n0 + ct * 16 + lo) * DD + kc * 32 + hi * 8;
            v16bf b = load_a_frag(bp);
            acc[ct] = __builtin_amdgcn_wmma_f32_16x16x32_bf16(
                false, a, false, b, (short)0, acc[ct], false, false);
        }
    }

#pragma unroll
    for (int ct = 0; ct < 4; ++ct) {
#pragma unroll
        for (int i = 0; i < 8; ++i) {
            int row = m0 + i + 8 * hi;
            int col = n0 + ct * 16 + lo;
            out[(size_t)row * H3 + col] = acc[ct][i];
        }
    }
}

// ----------------------- persistent recurrent kernel -------------------------
// 16 blocks total: blocks [0..7] forward, [8..15] backward; each owns a
// 64-column slice of H and computes all three gates for it. W_h slice
// (192 KB) staged into LDS in WMMA-fragment order. Per-step global barrier.
__global__ void gru_persistent_kernel(const float* __restrict__ xm,
                                      const unsigned short* __restrict__ whT,
                                      const float* __restrict__ b_fwd,
                                      const float* __restrict__ b_bwd,
                                      unsigned short* __restrict__ hbuf,
                                      unsigned* __restrict__ bars,
                                      float* __restrict__ out) {
    extern __shared__ unsigned short lds_wh[];   // 3*4*16*32*16 bf16 = 192 KB

    const int dir  = blockIdx.x >> 3;
    const int jblk = blockIdx.x & 7;
    const int j0   = jblk * 64;
    const int tid  = threadIdx.x;
    const int lane = tid & 31;
    const int wave = tid >> 5;
    const int lo   = lane & 15;
    const int hi   = lane >> 4;

    const unsigned short* Wh = whT + (size_t)dir * H3 * DD;
    const float* bias        = dir ? b_bwd : b_fwd;
    const float* xmD         = xm + (size_t)dir * MM * H3;
    unsigned short* hb       = hbuf + (size_t)dir * 2 * NB * HH;
    unsigned* bar            = bars + dir;

    // ---- Stage W_h slice into LDS in fragment order:
    // element base = (((g*4+ct)*16 + kc)*32 + lane)*16   (16 bf16 per lane)
    for (int c = tid; c < 12288; c += blockDim.x) {     // 12288 x 16B chunks
        int e8 = c & 1;
        int ln = (c >> 1) & 31;
        int kc = (c >> 6) & 15;
        int ct = (c >> 10) & 3;
        int g  = c >> 12;
        const unsigned short* src =
            Wh + (size_t)(g * HH + j0 + ct * 16 + (ln & 15)) * DD
               + kc * 32 + (ln >> 4) * 8 + e8 * 16;
        *(v8bf*)(lds_wh + (size_t)c * 8) = *(const v8bf*)src;
    }

    // ---- Zero h0 slice (buffer 0)
    for (int idx = tid; idx < 64 * 64; idx += blockDim.x) {
        int mm = idx >> 6;
        int jj = j0 + (idx & 63);
        hb[mm * HH + jj] = 0;
    }
    __syncthreads();

    // ---- global barrier: all 8 blocks of this direction staged + zeroed
    __threadfence();
    if (tid == 0) {
        atomicAdd(bar, 1u);
        while (__hip_atomic_load(bar, __ATOMIC_RELAXED, __HIP_MEMORY_SCOPE_AGENT) < 8u)
            __builtin_amdgcn_s_sleep(1);
    }
    __syncthreads();
    __threadfence();

    const int m0w = wave * 16;

    for (int t = 0; t < TT; ++t) {
        const int cur = t & 1;
        const unsigned short* hprev = hb + cur * NB * HH;
        unsigned short*       hnext = hb + (cur ^ 1) * NB * HH;

        // ---- H_lin = h_prev @ W_h[:, 3 gate slices], 12 WMMA tiles/wave
        v8f acc[3][4];
#pragma unroll
        for (int g = 0; g < 3; ++g)
#pragma unroll
            for (int ct = 0; ct < 4; ++ct) acc[g][ct] = (v8f)(0.0f);

        const unsigned short* arow = hprev + (size_t)(m0w + lo) * HH + hi * 8;
#pragma unroll 2
        for (int kc = 0; kc < HH / 32; ++kc) {
            v16bf a = load_a_frag(arow + kc * 32);
#pragma unroll
            for (int g = 0; g < 3; ++g) {
#pragma unroll
                for (int ct = 0; ct < 4; ++ct) {
                    const unsigned short* bp =
                        lds_wh + (size_t)((((g * 4 + ct) * 16 + kc) * 32) + lane) * 16;
                    v16bf b = *(const v16bf*)bp;
                    acc[g][ct] = __builtin_amdgcn_wmma_f32_16x16x32_bf16(
                        false, a, false, b, (short)0, acc[g][ct], false, false);
                }
            }
        }

        // ---- gates + state update + stores
        const int tout = dir ? (TT - 1 - t) : t;
#pragma unroll
        for (int ct = 0; ct < 4; ++ct) {
#pragma unroll
            for (int i = 0; i < 8; ++i) {
                int mrow = m0w + i + 8 * hi;            // batch index
                int jg   = j0 + ct * 16 + lo;           // hidden index
                const float* xr = xmD + ((size_t)mrow * TT + t) * H3;
                float z = sigmoidf(xr[jg]            + acc[0][ct][i] + bias[jg]);
                float r = sigmoidf(xr[HH + jg]       + acc[1][ct][i] + bias[HH + jg]);
                float g = tanhf   (xr[2 * HH + jg] + r * acc[2][ct][i] + bias[2 * HH + jg]);
                float hp = bf2f(hprev[mrow * HH + jg]);
                float h  = (1.0f - z) * g + z * hp;
                hnext[mrow * HH + jg] = f2bf(h);
                out[((size_t)mrow * TT + tout) * (2 * HH) + dir * HH + jg] = h;
            }
        }

        // ---- per-step grid barrier for this direction (monotonic counter)
        __threadfence();
        __syncthreads();
        if (tid == 0) {
            atomicAdd(bar, 1u);
            unsigned target = 8u * (unsigned)(t + 2);
            while (__hip_atomic_load(bar, __ATOMIC_RELAXED, __HIP_MEMORY_SCOPE_AGENT) < target)
                __builtin_amdgcn_s_sleep(1);
        }
        __syncthreads();
        __threadfence();
    }
}

// ------------------------------- launcher ------------------------------------
extern "C" void kernel_launch(void* const* d_in, const int* in_sizes, int n_in,
                              void* d_out, int out_size, void* d_ws, size_t ws_size,
                              hipStream_t stream) {
    const float* x   = (const float*)d_in[0];
    const float* Wxf = (const float*)d_in[1];
    const float* Whf = (const float*)d_in[2];
    const float* bf  = (const float*)d_in[3];
    const float* Wxb = (const float*)d_in[4];
    const float* Whb = (const float*)d_in[5];
    const float* bb  = (const float*)d_in[6];
    float* out = (float*)d_out;

    // Workspace layout (all 256B aligned)
    char* ws = (char*)d_ws;
    unsigned short* xbf = (unsigned short*)ws;                       // 32 MB
    size_t off = (size_t)MM * DD * 2;
    unsigned short* wxT = (unsigned short*)(ws + off);               // 3 MB
    off += (size_t)2 * H3 * DD * 2;
    unsigned short* whT = (unsigned short*)(ws + off);               // 3 MB
    off += (size_t)2 * H3 * DD * 2;
    float* xm = (float*)(ws + off);                                  // 384 MB
    off += (size_t)2 * MM * H3 * 4;
    unsigned short* hbuf = (unsigned short*)(ws + off);              // 256 KB
    off += (size_t)2 * 2 * NB * HH * 2;
    unsigned* bars = (unsigned*)(ws + off);

    init_barriers<<<1, 32, 0, stream>>>(bars);
    convert_x_kernel<<<(MM * DD) / (256 * 4), 256, 0, stream>>>(x, xbf);
    transpose_w_kernel<<<dim3((H3 * DD) / 256, 4), 256, 0, stream>>>(
        Wxf, Whf, Wxb, Whb, wxT, whT);
    proj_gemm_kernel<<<dim3(MM / 64, H3 / 64, 2), 128, 0, stream>>>(xbf, wxT, xm);
    gru_persistent_kernel<<<16, 128, 3 * 4 * 16 * 32 * 16 * 2, stream>>>(
        xm, whT, bf, bb, hbuf, bars, out);
}
